// XCA_72181220376984
// MI455X (gfx1250) — matrix operand order repaired and checked by reading, not compiled, not run on hardware
//
#include <hip/hip_runtime.h>
#include <hip/hip_bf16.h>
#include <math.h>

// ---------------- problem constants (from reference) ----------------
#define BB   8
#define NN   4096
#define CC   768
#define HH   12
#define DH   64
#define MROW (BB * NN)      // 32768
#define C3   (3 * CC)       // 2304
#define QKC  (2 * CC)       // 1536 (q,k columns)

typedef __attribute__((ext_vector_type(16))) __bf16          v16bf;
typedef __attribute__((ext_vector_type(8)))  __bf16          v8bf;
typedef __attribute__((ext_vector_type(16))) unsigned short  v16us;
typedef __attribute__((ext_vector_type(8)))  unsigned short  v8us;
typedef __attribute__((ext_vector_type(8)))  float           v8f;

// ---------------- helpers ----------------
__device__ __forceinline__ unsigned short f2bf(float f) {
    return __builtin_bit_cast(unsigned short, (__bf16)f);   // hw cvt (RNE)
}
__device__ __forceinline__ v8us pack8(v8f v) {
    return __builtin_bit_cast(v8us, __builtin_convertvector(v, v8bf));
}
__device__ __forceinline__ float bf2f(unsigned short h) {
    unsigned u = ((unsigned)h) << 16;
    return __builtin_bit_cast(float, u);
}
__device__ __forceinline__ v8f wmma_bf16(v16us a, v16us b, v8f c) {
    return __builtin_amdgcn_wmma_f32_16x16x32_bf16(
        false, __builtin_bit_cast(v16bf, a),
        false, __builtin_bit_cast(v16bf, b),
        (short)0, c, false, false);
}
__device__ __forceinline__ v16us frag_cat(v8us lo, v8us hi) {
    return __builtin_shufflevector(lo, hi, 0,1,2,3,4,5,6,7,8,9,10,11,12,13,14,15);
}
// gfx1250 async global->LDS copy (16B per lane) + wait (ASYNCcnt)
__device__ __forceinline__ void async_b128(unsigned lds_off, unsigned long long gaddr) {
    asm volatile("global_load_async_to_lds_b128 %0, %1, off"
                 :: "v"(lds_off), "v"(gaddr) : "memory");
}
__device__ __forceinline__ void wait_async0() {
    asm volatile("s_wait_asynccnt 0x0" ::: "memory");
}

// ---------------- one-time transforms ----------------
// x f32 -> bf16, vectorized (removes all conversion from GEMM hot loops)
__global__ __launch_bounds__(256) void cvt_x_kernel(
    const float* __restrict__ x, unsigned short* __restrict__ xbf) {
    size_t i = ((size_t)blockIdx.x * 256 + threadIdx.x) * 8;
    if (i >= (size_t)MROW * CC) return;
    *(v8us*)(xbf + i) = pack8(*(const v8f*)(x + i));
}
// W_qkv q/k part -> fragment-major: Wf[kt][ntile][lane][slot]
__global__ __launch_bounds__(256) void wqk_frag_kernel(
    const float* __restrict__ W, unsigned short* __restrict__ Wf) {
    int i = blockIdx.x * 256 + threadIdx.x;          // over 768*1536
    if (i >= CC * QKC) return;
    int k = i / QKC, col = i % QKC;
    int kt = k >> 5, kk = k & 31;
    int lane = ((kk >> 4) << 4) | (col & 15);
    int slot = kk & 15;
    size_t dst = ((((size_t)kt * (QKC / 16) + (col >> 4)) * 32) + lane) * 16 + slot;
    Wf[dst] = f2bf(W[(size_t)k * C3 + col]);
}
// transposed copy: dst[c][k] = bf16(src[k][colBase + c]),  c,k in [0,768)
__global__ __launch_bounds__(256) void wT_kernel(
    const float* __restrict__ src, int ld, int colBase, unsigned short* __restrict__ dst) {
    int i = blockIdx.x * 256 + threadIdx.x;          // over 768*768
    if (i >= CC * CC) return;
    int k = i / CC, c = i % CC;
    dst[(size_t)c * CC + k] = f2bf(src[(size_t)k * ld + colBase + c]);
}
__global__ __launch_bounds__(256) void zero_f32_kernel(float* __restrict__ p, int n) {
    int i = blockIdx.x * 256 + threadIdx.x;
    if (i < n) p[i] = 0.0f;
}

// ---------------- QK projection: qt[c][m] = (x @ Wqk)^T, c in [0,1536) ----------------
// grid (12, 256); block 256 (8 waves); A = x_bf via async double-buffered LDS
__global__ __launch_bounds__(256) void qk_gemm_kernel(
    const unsigned short* __restrict__ xbf, const unsigned short* __restrict__ Wf,
    unsigned short* __restrict__ qt) {
    __shared__ unsigned short As[2][128][40];        // 128 x 32 bf16 (+pad)

    const int tid = threadIdx.x, lane = tid & 31, wid = tid >> 5;
    const int waveR = wid >> 1, waveC = wid & 1;
    const int m_ = lane & 15, sel = lane >> 4;
    const int brow = blockIdx.y * 128;               // m
    const int bcol = blockIdx.x * 128;               // c
    const int ntile0 = (bcol >> 4) + waveC * 4;

    const int srow = tid >> 1, scs = (tid & 1) * 16;
    const unsigned short* gsrc = xbf + (size_t)(brow + srow) * CC + scs;

    v8f acc[2][4] = {};

    { // prologue: async stage k-tile 0 (32B per thread = 2 x b128)
        unsigned l = (unsigned)(size_t)&As[0][srow][scs];
        unsigned long long g = (unsigned long long)(size_t)gsrc;
        async_b128(l, g); async_b128(l + 16, g + 16);
    }
    for (int kt = 0; kt < CC / 32; ++kt) {
        wait_async0();
        __syncthreads();
        if (kt + 1 < CC / 32) {   // async stage next k-tile into other buffer
            unsigned l = (unsigned)(size_t)&As[(kt + 1) & 1][srow][scs];
            unsigned long long g = (unsigned long long)(size_t)(gsrc + (kt + 1) * 32);
            async_b128(l, g); async_b128(l + 16, g + 16);
        }
        const int buf = kt & 1;
        v16us afrag[2], bfrag[4];
        #pragma unroll
        for (int t = 0; t < 2; ++t) {
            int r = waveR * 32 + t * 16 + m_;
            afrag[t] = frag_cat(*(const v8us*)&As[buf][r][sel * 16],
                                *(const v8us*)&As[buf][r][sel * 16 + 8]);
        }
        #pragma unroll
        for (int u = 0; u < 4; ++u)
            bfrag[u] = *(const v16us*)(Wf + ((((size_t)kt * (QKC / 16)) + ntile0 + u) * 32 + lane) * 16);
        #pragma unroll
        for (int t = 0; t < 2; ++t)
            #pragma unroll
            for (int u = 0; u < 4; ++u)
                acc[t][u] = wmma_bf16(afrag[t], bfrag[u], acc[t][u]);
        __syncthreads();
    }
    // epilogue: transposed store, contiguous 16B per lane
    #pragma unroll
    for (int t = 0; t < 2; ++t) {
        #pragma unroll
        for (int u = 0; u < 4; ++u) {
            int c  = bcol + waveC * 64 + u * 16 + m_;
            int m0 = brow + waveR * 32 + t * 16 + 8 * sel;
            *(v8us*)(qt + (size_t)c * MROW + m0) = pack8(acc[t][u]);
        }
    }
}

// ---------------- V projection (transposed form, no LDS): vbuf[m][c] = x @ Wv ----------------
// D[c][m] = WvT[c][k] * xT[k][m]; grid (256, 6); block 256
__global__ __launch_bounds__(256) void v_gemm_kernel(
    const unsigned short* __restrict__ xbf, const unsigned short* __restrict__ WvT,
    unsigned short* __restrict__ vbuf) {
    const int tid = threadIdx.x, lane = tid & 31, wid = tid >> 5;
    const int waveR = wid >> 1, waveC = wid & 1;
    const int m_ = lane & 15, sel = lane >> 4;
    const int bm = blockIdx.x * 128;                 // m (cols of D)
    const int bc = blockIdx.y * 128;                 // c (rows of D)

    v8f acc[2][4] = {};
    for (int kt = 0; kt < CC / 32; ++kt) {
        int k0 = kt * 32;
        v16us afrag[2], bfrag[4];
        #pragma unroll
        for (int t = 0; t < 2; ++t) {                // A rows = c (weights)
            int c = bc + waveR * 32 + t * 16 + m_;
            const unsigned short* p = WvT + (size_t)c * CC + k0;
            afrag[t] = frag_cat(*(const v8us*)(p + sel * 16), *(const v8us*)(p + sel * 16 + 8));
        }
        #pragma unroll
        for (int u = 0; u < 4; ++u) {                // B cols = m (x rows, contiguous k)
            int m = bm + waveC * 64 + u * 16 + m_;
            const unsigned short* p = xbf + (size_t)m * CC + k0;
            __builtin_prefetch(p + 32, 0, 1);
            bfrag[u] = *(const v16us*)(p + sel * 16);
        }
        #pragma unroll
        for (int t = 0; t < 2; ++t)
            #pragma unroll
            for (int u = 0; u < 4; ++u)
                acc[t][u] = wmma_bf16(afrag[t], bfrag[u], acc[t][u]);
    }
    #pragma unroll
    for (int t = 0; t < 2; ++t) {
        #pragma unroll
        for (int u = 0; u < 4; ++u) {
            int m  = bm + waveC * 64 + u * 16 + m_;
            int c0 = bc + waveR * 32 + t * 16 + 8 * sel;
            *(v8us*)(vbuf + (size_t)m * CC + c0) = pack8(acc[t][u]);
        }
    }
}

// ---------------- inverse L2 norms over N: rows of qt ----------------
// grid (64, 2, B*H); block 256
__global__ __launch_bounds__(256) void norm_kernel(
    const unsigned short* __restrict__ qt, float* __restrict__ invq, float* __restrict__ invk) {
    const int d = blockIdx.x, s = blockIdx.y, bh = blockIdx.z;
    const int b = bh / HH, h = bh % HH;
    const unsigned short* p = qt + (size_t)(s * CC + h * DH + d) * MROW + (size_t)b * NN;

    float sum = 0.0f;
    int base = threadIdx.x * 16;
    v8us lo = *(const v8us*)(p + base);
    v8us hi = *(const v8us*)(p + base + 8);
    #pragma unroll
    for (int j = 0; j < 8; ++j) {
        float a = bf2f(lo[j]), c = bf2f(hi[j]);
        sum += a * a + c * c;
    }
    __shared__ float red[256];
    red[threadIdx.x] = sum;
    __syncthreads();
    for (int st = 128; st > 0; st >>= 1) {
        if (threadIdx.x < st) red[threadIdx.x] += red[threadIdx.x + st];
        __syncthreads();
    }
    if (threadIdx.x == 0) {
        float inv = 1.0f / fmaxf(sqrtf(red[0]), 1e-12f);
        float* dst = (s == 0) ? invq : invk;
        dst[(size_t)bh * DH + d] = inv;
    }
}

// ---------------- Gram: gram[bh] += Q Kt over an n-chunk (no LDS) ----------------
// grid (B*H, 8); block 256; wave = 16(d) x 32(e) tile pair
__global__ __launch_bounds__(256) void gram_kernel(
    const unsigned short* __restrict__ qt, float* __restrict__ gram) {
    const int bh = blockIdx.x, b = bh / HH, h = bh % HH;
    const int n0 = blockIdx.y * 512;
    const int tid = threadIdx.x, lane = tid & 31, wid = tid >> 5;
    const int m_ = lane & 15, sel = lane >> 4;
    const int d0 = (wid >> 1) * 16;
    const int e0b = (wid & 1) * 2;

    const unsigned short* qrow = qt + (size_t)(h * DH + d0 + m_) * MROW + (size_t)b * NN + n0;
    const unsigned short* krow0 = qt + (size_t)(CC + h * DH + (e0b + 0) * 16 + m_) * MROW + (size_t)b * NN + n0;
    const unsigned short* krow1 = qt + (size_t)(CC + h * DH + (e0b + 1) * 16 + m_) * MROW + (size_t)b * NN + n0;

    v8f acc[2] = {};
    for (int kk = 0; kk < 512; kk += 32) {
        __builtin_prefetch(qrow + kk + 128, 0, 1);
        __builtin_prefetch(krow0 + kk + 128, 0, 1);
        v16us a = frag_cat(*(const v8us*)(qrow + kk + sel * 16),
                           *(const v8us*)(qrow + kk + sel * 16 + 8));
        v16us b0 = *(const v16us*)(krow0 + kk + sel * 16);
        v16us b1 = *(const v16us*)(krow1 + kk + sel * 16);
        acc[0] = wmma_bf16(a, b0, acc[0]);
        acc[1] = wmma_bf16(a, b1, acc[1]);
    }
    float* g = gram + (size_t)bh * (DH * DH);
    #pragma unroll
    for (int u = 0; u < 2; ++u) {
        int col = (e0b + u) * 16 + m_;
        #pragma unroll
        for (int r = 0; r < 8; ++r) {
            int row = d0 + r + 8 * sel;
            atomicAdd(&g[row * DH + col], acc[u][r]);
        }
    }
}

// ---------------- scale + dropkey + softmax -> bf16 attn (row-major) ----------------
__global__ __launch_bounds__(64) void softmax_kernel(
    const float* __restrict__ gram, const float* __restrict__ invq,
    const float* __restrict__ invk, const float* __restrict__ temp,
    const unsigned char* __restrict__ mask, unsigned short* __restrict__ attn) {
    const int bh = blockIdx.x, h = bh % HH, d = threadIdx.x;
    const float iq = invq[(size_t)bh * DH + d] * temp[h];
    const float* grow = gram + (size_t)bh * (DH * DH) + d * DH;
    const unsigned char* mrow = mask + (size_t)bh * (DH * DH) + d * DH;
    const float* ik = invk + (size_t)bh * DH;

    float vals[DH];
    float mx = -INFINITY;
    #pragma unroll 8
    for (int e = 0; e < DH; ++e) {
        float v = grow[e] * iq * ik[e];
        if (mrow[e]) v -= 1.0e12f;
        vals[e] = v;
        mx = fmaxf(mx, v);
    }
    float sum = 0.0f;
    #pragma unroll 8
    for (int e = 0; e < DH; ++e) { float ex = expf(vals[e] - mx); vals[e] = ex; sum += ex; }
    float inv = 1.0f / sum;
    unsigned short* arow = attn + (size_t)bh * (DH * DH) + d * DH;
    #pragma unroll
    for (int j0 = 0; j0 < DH; j0 += 8) {
        v8f o;
        #pragma unroll
        for (int j = 0; j < 8; ++j) o[j] = vals[j0 + j] * inv;
        *(v8us*)(arow + j0) = pack8(o);
    }
}

// ---------------- AV: aout[m][h*64+d] = attn[d][e] * vbuf[m][h*64+e] (no LDS) ----------------
// grid (B*H, 16); block 256; wave = 64(d) x 32(n)
__global__ __launch_bounds__(256) void av_kernel(
    const unsigned short* __restrict__ vbuf, const unsigned short* __restrict__ attn,
    unsigned short* __restrict__ aout) {
    const int bh = blockIdx.x, b = bh / HH, h = bh % HH;
    const int n0 = blockIdx.y * 256;
    const int tid = threadIdx.x, lane = tid & 31, wid = tid >> 5;
    const int m_ = lane & 15, sel = lane >> 4;

    const unsigned short* ap = attn + (size_t)bh * (DH * DH);

    v8f acc[4][2] = {};
    #pragma unroll
    for (int e0 = 0; e0 < DH; e0 += 32) {
        v16us a[4], bf[2];
        #pragma unroll
        for (int t = 0; t < 4; ++t) {
            const unsigned short* p = ap + (t * 16 + m_) * DH + e0;
            a[t] = frag_cat(*(const v8us*)(p + sel * 16), *(const v8us*)(p + sel * 16 + 8));
        }
        #pragma unroll
        for (int u = 0; u < 2; ++u) {
            int n = n0 + (wid * 2 + u) * 16 + m_;
            const unsigned short* p = vbuf + (size_t)(b * NN + n) * CC + h * DH + e0;
            bf[u] = *(const v16us*)(p + sel * 16);
        }
        #pragma unroll
        for (int t = 0; t < 4; ++t)
            #pragma unroll
            for (int u = 0; u < 2; ++u)
                acc[t][u] = wmma_bf16(a[t], bf[u], acc[t][u]);
    }
    #pragma unroll
    for (int t = 0; t < 4; ++t) {
        #pragma unroll
        for (int u = 0; u < 2; ++u) {
            int n = n0 + (wid * 2 + u) * 16 + m_;
            int d0 = t * 16 + 8 * sel;
            *(v8us*)(aout + (size_t)(b * NN + n) * CC + h * DH + d0) = pack8(acc[t][u]);
        }
    }
}

// ---------------- Proj (transposed form, no LDS): out[m][c] = aout @ Wp + bias + x ----------------
// D[c][m] = WpT[c][k] * aoutT[k][m]; grid (256, 6); block 256
__global__ __launch_bounds__(256) void proj_gemm_kernel(
    const unsigned short* __restrict__ aout, const unsigned short* __restrict__ WpT,
    const float* __restrict__ bias, const float* __restrict__ x,
    float* __restrict__ out) {
    const int tid = threadIdx.x, lane = tid & 31, wid = tid >> 5;
    const int waveR = wid >> 1, waveC = wid & 1;
    const int m_ = lane & 15, sel = lane >> 4;
    const int bm = blockIdx.x * 128;                 // m (cols of D)
    const int bc = blockIdx.y * 128;                 // c (rows of D)

    v8f acc[2][4] = {};
    for (int kt = 0; kt < CC / 32; ++kt) {
        int k0 = kt * 32;
        v16us afrag[2], bfrag[4];
        #pragma unroll
        for (int t = 0; t < 2; ++t) {
            int c = bc + waveR * 32 + t * 16 + m_;
            const unsigned short* p = WpT + (size_t)c * CC + k0;
            afrag[t] = frag_cat(*(const v8us*)(p + sel * 16), *(const v8us*)(p + sel * 16 + 8));
        }
        #pragma unroll
        for (int u = 0; u < 4; ++u) {
            int m = bm + waveC * 64 + u * 16 + m_;
            const unsigned short* p = aout + (size_t)m * CC + k0;
            __builtin_prefetch(p + 32, 0, 1);
            bfrag[u] = *(const v16us*)(p + sel * 16);
        }
        #pragma unroll
        for (int t = 0; t < 2; ++t)
            #pragma unroll
            for (int u = 0; u < 4; ++u)
                acc[t][u] = wmma_bf16(afrag[t], bfrag[u], acc[t][u]);
    }
    #pragma unroll
    for (int t = 0; t < 2; ++t) {
        #pragma unroll
        for (int u = 0; u < 4; ++u) {
            int m  = bm + waveC * 64 + u * 16 + m_;
            int c0 = bc + waveR * 32 + t * 16 + 8 * sel;
            v8f bi = *(const v8f*)(bias + c0);
            v8f rx = *(const v8f*)(x + (size_t)m * CC + c0);
            v8f o;
            #pragma unroll
            for (int r = 0; r < 8; ++r) o[r] = acc[t][u][r] + bi[r] + rx[r];
            *(v8f*)(out + (size_t)m * CC + c0) = o;
        }
    }
}

// ---------------- host launch ----------------
extern "C" void kernel_launch(void* const* d_in, const int* in_sizes, int n_in,
                              void* d_out, int out_size, void* d_ws, size_t ws_size,
                              hipStream_t stream) {
    const float*         x      = (const float*)d_in[0];
    const float*         W_qkv  = (const float*)d_in[1];
    const float*         W_proj = (const float*)d_in[2];
    const float*         b_proj = (const float*)d_in[3];
    const float*         temp   = (const float*)d_in[4];
    const unsigned char* dmask  = (const unsigned char*)d_in[5];
    float*               out    = (float*)d_out;
    (void)in_sizes; (void)n_in; (void)out_size; (void)ws_size;

    char* ws = (char*)d_ws;
    size_t off = 0;
    auto carve = [&](size_t bytes) { void* p = ws + off; off += (bytes + 255) & ~(size_t)255; return p; };
    unsigned short* xbf     = (unsigned short*)carve((size_t)MROW * CC * 2);    // x bf16 [m][c]
    unsigned short* qt      = (unsigned short*)carve((size_t)QKC * MROW * 2);   // q,k transposed [c][m]
    unsigned short* vbuf    = (unsigned short*)carve((size_t)MROW * CC * 2);    // v [m][c]
    unsigned short* aout    = (unsigned short*)carve((size_t)MROW * CC * 2);    // attn@v [m][c]
    float*          gram    = (float*)carve((size_t)BB * HH * DH * DH * 4);
    unsigned short* attn_bf = (unsigned short*)carve((size_t)BB * HH * DH * DH * 2);
    float*          invq    = (float*)carve((size_t)BB * HH * DH * 4);
    float*          invk    = (float*)carve((size_t)BB * HH * DH * 4);
    unsigned short* Wf_qk   = (unsigned short*)carve((size_t)(CC / 32) * (QKC / 16) * 32 * 16 * 2);
    unsigned short* WvT     = (unsigned short*)carve((size_t)CC * CC * 2);
    unsigned short* WpT     = (unsigned short*)carve((size_t)CC * CC * 2);

    // one-time transforms + gram zeroing
    cvt_x_kernel<<<(MROW * CC / 8 + 255) / 256, 256, 0, stream>>>(x, xbf);
    wqk_frag_kernel<<<(CC * QKC + 255) / 256, 256, 0, stream>>>(W_qkv, Wf_qk);
    wT_kernel<<<(CC * CC + 255) / 256, 256, 0, stream>>>(W_qkv, C3, 2 * CC, WvT);
    wT_kernel<<<(CC * CC + 255) / 256, 256, 0, stream>>>(W_proj, CC, 0, WpT);
    zero_f32_kernel<<<(BB * HH * DH * DH + 255) / 256, 256, 0, stream>>>(gram, BB * HH * DH * DH);

    // q,k projection (transposed output, async double-buffered LDS for x)
    qk_gemm_kernel<<<dim3(QKC / 128, MROW / 128), 256, 0, stream>>>(xbf, Wf_qk, qt);
    // v projection ([m][c] output, transposed formulation, no LDS)
    v_gemm_kernel<<<dim3(MROW / 128, CC / 128), 256, 0, stream>>>(xbf, WvT, vbuf);
    // inverse L2 norms over N
    norm_kernel<<<dim3(DH, 2, BB * HH), 256, 0, stream>>>(qt, invq, invk);
    // Gram accumulation
    gram_kernel<<<dim3(BB * HH, 8), 256, 0, stream>>>(qt, gram);
    // softmax with temperature + dropkey
    softmax_kernel<<<BB * HH, 64, 0, stream>>>(gram, invq, invk, temp, dmask, attn_bf);
    // attn @ V
    av_kernel<<<dim3(BB * HH, NN / 256), 256, 0, stream>>>(vbuf, attn_bf, aout);
    // output projection + bias + residual
    proj_gemm_kernel<<<dim3(MROW / 128, CC / 128), 256, 0, stream>>>(aout, WpT, b_proj, x, out);
}
